// MillionsMoE_4947802325414
// MI455X (gfx1250) — matrix operand order repaired and checked by reading, not compiled, not run on hardware
//
#include <hip/hip_runtime.h>
#include <hip/hip_bf16.h>
#include <math.h>

typedef __attribute__((ext_vector_type(16))) _Float16 v16h;
typedef __attribute__((ext_vector_type(8)))  _Float16 v8h;
typedef __attribute__((ext_vector_type(8)))  float    v8f;

constexpr int NT   = 4096;   // tokens (2*2048)
constexpr int D    = 512;    // d_model
constexpr int H    = 8;      // heads
constexpr int DK   = 128;    // per-head query dim
constexpr int HALF = 64;     // half of d_keys
constexpr int HD   = H * DK; // 1024
constexpr int E    = 512;    // sub-experts per dim
constexpr int TK   = 8;      // top_k

#define SHUF16(a, b) __builtin_shufflevector(a, b, 0,1,2,3,4,5,6,7,8,9,10,11,12,13,14,15)

// ---------------- Kernel 1: f32 -> f16 pack ----------------
__global__ void k_f32tof16(const float* __restrict__ s, _Float16* __restrict__ d, int n) {
  int i = blockIdx.x * blockDim.x + threadIdx.x;
  int stride = gridDim.x * blockDim.x;
  for (; i < n; i += stride) d[i] = (_Float16)s[i];
}

// ---------------- Kernel 2: qh = q @ Wq^T + bq (WMMA f16) ----------------
// one wave = one 16x16 output tile; waves = 256 n-tiles * 64 o-tiles
__global__ void k_qcast(const _Float16* __restrict__ q, const _Float16* __restrict__ Wq,
                        const float* __restrict__ bq, _Float16* __restrict__ qh) {
  int wave = blockIdx.x * (blockDim.x >> 5) + (threadIdx.x >> 5);
  int lane = threadIdx.x & 31;
  int n_tile = wave >> 6;
  int o_tile = wave & 63;
  int lo = lane & 15, hi = lane >> 4;

  const _Float16* arow = q  + (size_t)(n_tile * 16 + lo) * D;
  const _Float16* brow = Wq + (size_t)(o_tile * 16 + lo) * D;

  v8f acc = {};
  for (int kb = 0; kb < D; kb += 32) {
    // A fragment (16x32 f16): lanes 0-15 K=g..g+7,16+g..; g = hi*8
    v8h a0 = *(const v8h*)(arow + kb + hi * 8);
    v8h a1 = *(const v8h*)(arow + kb + 16 + hi * 8);
    v16h A = SHUF16(a0, a1);
    // B fragment (32x16 f16): lane N = lo, K = hi*8 + e, 16 + hi*8 + e (same as A)
    v8h b0 = *(const v8h*)(brow + kb + hi * 8);
    v8h b1 = *(const v8h*)(brow + kb + 16 + hi * 8);
    v16h B = SHUF16(b0, b1);
    acc = __builtin_amdgcn_wmma_f32_16x16x32_f16(false, A, false, B, (short)0, acc,
                                                 false, false);
  }
  int col = o_tile * 16 + lo;
  float bias = bq[col];
  int rbase = n_tile * 16 + hi * 8;
#pragma unroll
  for (int v = 0; v < 8; ++v)
    qh[(size_t)(rbase + v) * HD + col] = (_Float16)(acc[v] + bias);
}

// ------- Kernel 3: sub-key scores (WMMA) + 2-stage wave top-k + softmax -------
// block = 256 threads (8 waves), one block per (16-token tile, head)
__global__ void k_scores_topk(const _Float16* __restrict__ qh,
                              const _Float16* __restrict__ keys,
                              int* __restrict__ tidx, float* __restrict__ tw) {
  __shared__ float sc[16][E + 4];
  __shared__ float topS[16][2][TK];
  __shared__ int   topI[16][2][TK];

  int n_tile = blockIdx.x >> 3;
  int h      = blockIdx.x & 7;
  int wave = threadIdx.x >> 5, lane = threadIdx.x & 31;
  int lo = lane & 15, hi = lane >> 4;

  for (int c = 0; c < 2; ++c) {
    // A fragments (shared across all 32 N-tiles): qh[:, h*128 + c*64 + 0..63]
    const _Float16* arow = qh + (size_t)(n_tile * 16 + lo) * HD + h * DK + c * HALF;
    v16h A0 = SHUF16(*(const v8h*)(arow + hi * 8),      *(const v8h*)(arow + 16 + hi * 8));
    v16h A1 = SHUF16(*(const v8h*)(arow + 32 + hi * 8), *(const v8h*)(arow + 48 + hi * 8));

    const _Float16* kc = keys + (size_t)(h * 2 + c) * E * HALF;
#pragma unroll
    for (int t = 0; t < 4; ++t) {
      int nt = wave * 4 + t;                       // N tile 0..31
      const _Float16* brow = kc + (size_t)(nt * 16 + lo) * HALF;
      v16h B0 = SHUF16(*(const v8h*)(brow + hi * 8),
                       *(const v8h*)(brow + 16 + hi * 8));
      v16h B1 = SHUF16(*(const v8h*)(brow + 32 + hi * 8),
                       *(const v8h*)(brow + 48 + hi * 8));
      v8f acc = {};
      acc = __builtin_amdgcn_wmma_f32_16x16x32_f16(false, A0, false, B0, (short)0, acc,
                                                   false, false);
      acc = __builtin_amdgcn_wmma_f32_16x16x32_f16(false, A1, false, B1, (short)0, acc,
                                                   false, false);
      int e0 = nt * 16 + lo;
#pragma unroll
      for (int v = 0; v < 8; ++v) sc[hi * 8 + v][e0] = acc[v];
    }
    __syncthreads();

    // per-half top-8 of 512: wave handles tokens (wave) and (wave+8)
    for (int rep = 0; rep < 2; ++rep) {
      int t = wave + rep * 8;
      float v[16];
#pragma unroll
      for (int j = 0; j < 16; ++j) v[j] = sc[t][lane + 32 * j];
      for (int r = 0; r < TK; ++r) {
        float bv = -1e30f; int bj = 0;
#pragma unroll
        for (int j = 0; j < 16; ++j) if (v[j] > bv) { bv = v[j]; bj = j; }
        int bidx = lane + 32 * bj;
        for (int off = 16; off > 0; off >>= 1) {
          float ov = __shfl_xor(bv, off, 32);
          int   oi = __shfl_xor(bidx, off, 32);
          if (ov > bv) { bv = ov; bidx = oi; }
        }
        if ((bidx & 31) == lane) v[bidx >> 5] = -1e30f;   // mask winner
        if (lane == 0) { topS[t][c][r] = bv; topI[t][c][r] = bidx; }
      }
    }
    __syncthreads();
  }

  // combine: 8x8 pair sums -> top-8 of 64 -> softmax weights
  for (int rep = 0; rep < 2; ++rep) {
    int t = wave + rep * 8;
    float pv[2]; int pi[2];
#pragma unroll
    for (int s = 0; s < 2; ++s) {
      int p = lane + 32 * s;
      int i0 = p >> 3, i1 = p & 7;
      pv[s] = topS[t][0][i0] + topS[t][1][i1];
      pi[s] = topI[t][0][i0] * E + topI[t][1][i1];
    }
    float m = 0.f, sum = 0.f, mye = 0.f; int myi = 0;
    for (int r = 0; r < TK; ++r) {
      float bv; int slot;
      if (pv[0] >= pv[1]) { bv = pv[0]; slot = 0; } else { bv = pv[1]; slot = 1; }
      int bidx = pi[slot];
      int bp   = lane + 32 * slot;
      for (int off = 16; off > 0; off >>= 1) {
        float ov = __shfl_xor(bv, off, 32);
        int   oi = __shfl_xor(bidx, off, 32);
        int   op = __shfl_xor(bp, off, 32);
        if (ov > bv) { bv = ov; bidx = oi; bp = op; }
      }
      if ((bp & 31) == lane) pv[bp >> 5] = -1e30f;
      if (r == 0) m = bv;                      // rounds are descending -> max
      float er = __expf(bv - m);
      sum += er;
      if (lane == r) { mye = er; myi = bidx; }
    }
    int tok = n_tile * 16 + t;
    if (lane < TK) {
      tidx[(size_t)(tok * H + h) * TK + lane] = myi;
      tw  [(size_t)(tok * H + h) * TK + lane] = mye / sum;
    }
  }
}

// ---------------- Kernel 4: vals = gelu(q . w_down[idx]) * softmax_w ----------------
// one wave per (token, h*K+k)
__global__ void k_acts(const float* __restrict__ q, const float* __restrict__ wdown,
                       const int* __restrict__ tidx, const float* __restrict__ tw,
                       float* __restrict__ vals) {
  int gid  = blockIdx.x * (blockDim.x >> 5) + (threadIdx.x >> 5);
  int lane = threadIdx.x & 31;
  int n = gid >> 6, hk = gid & 63;
  int e = tidx[(size_t)n * 64 + hk];
  const float* qr = q + (size_t)n * D;
  const float* wr = wdown + (size_t)e * D;
  float s = 0.f;
  for (int j = lane; j < D; j += 32) s += qr[j] * wr[j];
  for (int off = 16; off > 0; off >>= 1) s += __shfl_xor(s, off, 32);
  if (lane == 0) {
    float x = s;
    float g = 0.5f * x * (1.0f + tanhf(0.7978845608f * (x + 0.044715f * x * x * x)));
    vals[(size_t)n * 64 + hk] = g * tw[(size_t)n * 64 + hk];
  }
}

// ---------------- Kernel 5: out[n,:] = sum_hk vals * w_up[idx] ----------------
// one block per token; 256 threads each own 2 output dims; prefetch next row
__global__ void k_out(const float* __restrict__ wup, const int* __restrict__ tidx,
                      const float* __restrict__ vals, float* __restrict__ out) {
  __shared__ float sv[64];
  __shared__ int   si[64];
  int n = blockIdx.x, tid = threadIdx.x;
  if (tid < 64) {
    sv[tid] = vals[(size_t)n * 64 + tid];
    si[tid] = tidx[(size_t)n * 64 + tid];
  }
  __syncthreads();
  float a0 = 0.f, a1 = 0.f;
  for (int e = 0; e < 64; ++e) {
    const float* wr = wup + (size_t)si[e] * D;
    if (e + 1 < 64)
      __builtin_prefetch(wup + (size_t)si[e + 1] * D + tid, 0, 0);  // global_prefetch_b8
    float vv = sv[e];
    a0 += vv * wr[tid];
    a1 += vv * wr[tid + 256];
  }
  out[(size_t)n * D + tid]       = a0;
  out[(size_t)n * D + tid + 256] = a1;
}

extern "C" void kernel_launch(void* const* d_in, const int* in_sizes, int n_in,
                              void* d_out, int out_size, void* d_ws, size_t ws_size,
                              hipStream_t stream) {
  (void)in_sizes; (void)n_in; (void)out_size; (void)ws_size;
  const float* queries = (const float*)d_in[0];   // [NT, D]
  const float* Wq      = (const float*)d_in[1];   // [HD, D]
  const float* bq      = (const float*)d_in[2];   // [HD]
  const float* keys    = (const float*)d_in[3];   // [H, 2, E, HALF]
  const float* wdown   = (const float*)d_in[4];   // [E*E, D]
  const float* wup     = (const float*)d_in[5];   // [E*E, D]
  float* out = (float*)d_out;

  // workspace carve-up (~18 MB)
  _Float16* q16   = (_Float16*)d_ws;
  _Float16* Wq16  = q16   + (size_t)NT * D;
  _Float16* k16   = Wq16  + (size_t)HD * D;
  _Float16* qh16  = k16   + (size_t)H * 2 * E * HALF;
  char*     p2    = (char*)(qh16 + (size_t)NT * HD);
  int*      tidx  = (int*)p2;
  float*    tw    = (float*)(tidx + (size_t)NT * 64);
  float*    vals  = tw + (size_t)NT * 64;

  k_f32tof16<<<1024, 256, 0, stream>>>(queries, q16, NT * D);
  k_f32tof16<<<512, 256, 0, stream>>>(Wq, Wq16, HD * D);
  k_f32tof16<<<512, 256, 0, stream>>>(keys, k16, H * 2 * E * HALF);

  // 256 n-tiles * 64 o-tiles = 16384 waves / 8 waves per block
  k_qcast<<<2048, 256, 0, stream>>>(q16, Wq16, bq, qh16);

  // one block per (16-token tile, head)
  k_scores_topk<<<(NT / 16) * H, 256, 0, stream>>>(qh16, k16, tidx, tw);

  // one wave per (token, expert-slot): NT*64 waves / 8 per block
  k_acts<<<(NT * 64) / 8, 256, 0, stream>>>(queries, wdown, tidx, tw, vals);

  // one block per token
  k_out<<<NT, 256, 0, stream>>>(wup, tidx, vals, out);
}
